// GatedDeltaNetLayer_65094524338663
// MI455X (gfx1250) — matrix operand order repaired and checked by reading, not compiled, not run on hardware
//
#include <hip/hip_runtime.h>

// ---------------------------------------------------------------------------
// Gated DeltaNet, chunkwise-parallel form for CDNA5 (gfx1250, wave32, WMMA).
// One workgroup per (b,h); state S [128x128] f32 lives in LDS with a bf16
// shadow for WMMA operands. Chunk C=32 so the K^T@U inner dim == 32 and maps
// 1:1 onto v_wmma_f32_16x16x32_bf16. Register budget targets 4 waves/SIMD
// (2 workgroups/WGP, matching the ~145KB LDS footprint).
// ---------------------------------------------------------------------------

typedef __attribute__((ext_vector_type(16))) __bf16 v16bf;
typedef __attribute__((ext_vector_type(8)))  float  v8f;

union ABf16 { unsigned int u32[8]; v16bf v; };
union CF32  { float f[8]; v8f v; };

__device__ __forceinline__ unsigned short f2bf(float x) {
    unsigned int u = __float_as_uint(x);
    u += 0x7FFFu + ((u >> 16) & 1u);   // round-to-nearest-even
    return (unsigned short)(u >> 16);
}

#define TC 32
#define KD 128
#define VD 128

__global__ __launch_bounds__(256)
__attribute__((amdgpu_waves_per_eu(4)))
void gated_delta_chunk_kernel(const float* __restrict__ kg,
                              const float* __restrict__ vg,
                              const float* __restrict__ betag,
                              const float* __restrict__ gg,
                              const float* __restrict__ s0g,
                              float* __restrict__ outg,
                              int B, int T, int H)
{
    extern __shared__ unsigned char smem[];
    float*          Sf   = (float*)(smem);                     // [v][k] 64KB (f32 state, transposed)
    unsigned short* Sbt  = (unsigned short*)(smem + 65536);    // [v][k] 32KB (bf16 shadow)
    unsigned short* Kb   = (unsigned short*)(smem + 98304);    // [t][k]  8KB
    unsigned short* Kbt  = (unsigned short*)(smem + 106496);   // [k][t]  8KB
    unsigned short* Ubt  = (unsigned short*)(smem + 114688);   // [v][t]  8KB (w_i * err, bf16)
    float*          P0   = (float*)(smem + 122880);            // [t][v] 16KB
    float*          Amat = (float*)(smem + 139264);            // [32][32]
    float*          Lmat = (float*)(smem + 143360);            // [32][32]
    float*          g_s    = (float*)(smem + 147456);
    float*          beta_s = g_s + 32;
    float*          cum    = g_s + 64;    // inclusive prefix of log(g)
    float*          Gp     = g_s + 96;    // G_{t-1}
    float*          wfac   = g_s + 128;   // (G_C/G_t)*beta_t
    float*          dfac   = g_s + 160;   // beta_t*||k_t||^2

    const int tid  = threadIdx.x;
    const int wave = tid >> 5;
    const int lane = tid & 31;
    const int ln16 = lane & 15;
    const int hf   = lane >> 4;

    const int bh = blockIdx.x;
    const int b  = bh / H;
    const int h  = bh % H;

    const long strideT = (long)H * KD;                      // elems between steps
    const float* kbh = kg + (long)b * T * H * KD + (long)h * KD;
    const float* vbh = vg + (long)b * T * H * KD + (long)h * KD;
    float*       obh = outg + (long)b * T * H * VD + (long)h * VD;
    const float* bbh = betag + (long)b * T * H + h;
    const float* gbh = gg    + (long)b * T * H + h;

    // ---- init state (transposed into LDS) ----
    for (int idx = tid; idx < KD * VD; idx += 256) {
        int kk = idx >> 7, vv = idx & 127;
        float s = s0g[((long)bh * KD + kk) * VD + vv];
        Sf[vv * KD + kk]  = s;
        Sbt[vv * KD + kk] = f2bf(s);
    }
    __syncthreads();

    const int nChunks = T / TC;
    for (int c = 0; c < nChunks; ++c) {
        const int t0 = c * TC;

        // ---------------- Phase A: load K chunk, gates ----------------
        for (int idx = tid; idx < TC * KD; idx += 256) {
            int t = idx >> 7, kk = idx & 127;
            const float* src = kbh + (long)(t0 + t) * strideT + kk;
            float kv = *src;
            if (c + 1 < nChunks) __builtin_prefetch(src + TC * strideT, 0, 1);
            unsigned short kb = f2bf(kv);
            Kb[t * KD + kk]  = kb;
            Kbt[kk * TC + t] = kb;
        }
        if (tid < TC) {
            g_s[tid]    = gbh[(long)(t0 + tid) * H];
            beta_s[tid] = bbh[(long)(t0 + tid) * H];
        }
        __syncthreads();
        if (tid == 0) {                      // serial 32-step log prefix (trivial)
            float acc = 0.f;
            for (int t = 0; t < TC; ++t) { acc += __logf(g_s[t]); cum[t] = acc; }
        }

        // ---------------- Phase B: P0 = K@S0 ; Amat = K@K^T -----------
        {   // GEMM1: wave w owns v-tile tc=w, both t-tiles
            const int tc = wave;
            for (int tr = 0; tr < 2; ++tr) {
                CF32 acc;
                #pragma unroll
                for (int r = 0; r < 8; ++r) acc.f[r] = 0.f;
                #pragma unroll
                for (int ks = 0; ks < 4; ++ks) {
                    ABf16 a, bm;
                    #pragma unroll
                    for (int r = 0; r < 8; ++r) {
                        int kl = 16 * hf + 2 * r;   // A frag K index
                        a.u32[r]  = *(const unsigned int*)&Kb [(16 * tr + ln16) * KD + ks * 32 + kl];
                        bm.u32[r] = *(const unsigned int*)&Sbt[(16 * tc + ln16) * KD + ks * 32 + 16 * hf + 2 * r];
                    }
                    acc.v = __builtin_amdgcn_wmma_f32_16x16x32_bf16(
                                false, a.v, false, bm.v, (short)0, acc.v, false, false);
                }
                #pragma unroll
                for (int r = 0; r < 8; ++r)
                    P0[(16 * tr + r + 8 * hf) * VD + 16 * tc + ln16] = acc.f[r];
            }
        }
        if (wave < 4) {  // GEMM2: 4 tiles of [32,32] = K@K^T
            const int tr = wave >> 1, tc = wave & 1;
            CF32 acc;
            #pragma unroll
            for (int r = 0; r < 8; ++r) acc.f[r] = 0.f;
            #pragma unroll
            for (int ks = 0; ks < 4; ++ks) {
                ABf16 a, bm;
                #pragma unroll
                for (int r = 0; r < 8; ++r) {
                    int kl = 16 * hf + 2 * r;
                    a.u32[r]  = *(const unsigned int*)&Kb[(16 * tr + ln16) * KD + ks * 32 + kl];
                    bm.u32[r] = *(const unsigned int*)&Kb[(16 * tc + ln16) * KD + ks * 32 + 16 * hf + 2 * r];
                }
                acc.v = __builtin_amdgcn_wmma_f32_16x16x32_bf16(
                            false, a.v, false, bm.v, (short)0, acc.v, false, false);
            }
            #pragma unroll
            for (int r = 0; r < 8; ++r)
                Amat[(16 * tr + r + 8 * hf) * TC + 16 * tc + ln16] = acc.f[r];
        }
        __syncthreads();

        // ---------------- decay factors + L matrix --------------------
        if (tid < TC) {
            int t = tid;
            Gp[t]   = (t == 0) ? 1.f : __expf(cum[t - 1]);
            wfac[t] = __expf(cum[TC - 1] - cum[t]) * beta_s[t];
            dfac[t] = beta_s[t] * Amat[t * TC + t];
        }
        for (int idx = tid; idx < TC * TC; idx += 256) {
            int t = idx >> 5, i = idx & 31;
            Lmat[idx] = (i < t) ? __expf(cum[t - 1] - cum[i]) * beta_s[i] * Amat[idx] : 0.f;
        }
        __syncthreads();

        // -------- Phase C: forward substitution (thread = v column) ---
        // Only e[0..31] stays live across the unrolled scan; v is loaded
        // per-step (coalesced) to keep VGPR usage under the 4-waves/SIMD cap.
        if (tid < VD) {
            const int v = tid;
            const float* vrow = vbh + (long)t0 * strideT + v;
            float* orow = obh + (long)t0 * strideT + v;
            float e[TC];
            #pragma unroll
            for (int t = 0; t < TC; ++t) {
                float vv = vrow[(long)t * strideT];
                float pred = Gp[t] * P0[t * VD + v];
                #pragma unroll
                for (int i = 0; i < t; ++i) pred += Lmat[t * TC + i] * e[i];
                float et = vv - pred;
                e[t] = et;
                orow[(long)t * strideT] = g_s[t] * pred + dfac[t] * et;   // out_t
                Ubt[v * TC + t] = f2bf(wfac[t] * et);
            }
        }
        __syncthreads();

        // -------- Phase D: S = G_C*S + K^T @ U  (C preloaded w/ G_C*S) -
        {
            const float GC = __expf(cum[TC - 1]);
            const int tr = wave;                     // k-dim strip
            for (int tc = 0; tc < 8; ++tc) {         // v-dim tiles (no forced unroll)
                ABf16 a, bm;
                #pragma unroll
                for (int r = 0; r < 8; ++r) {
                    int il = 16 * hf + 2 * r;   // inner = timestep
                    a.u32[r]  = *(const unsigned int*)&Kbt[(16 * tr + ln16) * TC + il];
                    bm.u32[r] = *(const unsigned int*)&Ubt[(16 * tc + ln16) * TC + 16 * hf + 2 * r];
                }
                CF32 acc;
                #pragma unroll
                for (int r = 0; r < 8; ++r)
                    acc.f[r] = GC * Sf[(16 * tc + ln16) * KD + 16 * tr + r + 8 * hf];
                acc.v = __builtin_amdgcn_wmma_f32_16x16x32_bf16(
                            false, a.v, false, bm.v, (short)0, acc.v, false, false);
                #pragma unroll
                for (int r = 0; r < 8; ++r) {
                    float d = acc.f[r];
                    Sf [(16 * tc + ln16) * KD + 16 * tr + r + 8 * hf] = d;
                    Sbt[(16 * tc + ln16) * KD + 16 * tr + r + 8 * hf] = f2bf(d);
                }
            }
        }
        __syncthreads();
    }

    // ---- final state (second tuple output), [B,H,K,V] ----
    float* fs = outg + (long)B * T * H * VD;
    for (int idx = tid; idx < KD * VD; idx += 256) {
        int kk = idx >> 7, vv = idx & 127;
        fs[((long)bh * KD + kk) * VD + vv] = Sf[vv * KD + kk];
    }
}

extern "C" void kernel_launch(void* const* d_in, const int* in_sizes, int n_in,
                              void* d_out, int out_size, void* d_ws, size_t ws_size,
                              hipStream_t stream) {
    const float* k    = (const float*)d_in[0];
    const float* v    = (const float*)d_in[1];
    const float* beta = (const float*)d_in[2];
    const float* g    = (const float*)d_in[3];
    const float* s0   = (const float*)d_in[4];
    float* out = (float*)d_out;

    const int BH = in_sizes[4] / (KD * VD);   // B*H from initial_state
    const int H  = 16;
    const int B  = BH / H;
    const int T  = in_sizes[2] / BH;          // beta is [B,T,H]

    const size_t smem = 148480;               // ~145 KB: 2 workgroups / WGP
    gated_delta_chunk_kernel<<<dim3(BH), dim3(256), smem, stream>>>(
        k, v, beta, g, s0, out, B, T, H);
}
